// GlobalLossD_10557029614243
// MI455X (gfx1250) — matrix operand order repaired and checked by reading, not compiled, not run on hardware
//
#include <hip/hip_runtime.h>

// ---- problem constants (match reference) ----
#define N_TOTAL 384          // rows of reg_pred
#define D_FEAT  131072       // features (2^17)
#define BS      128          // N_PARTS*N_DATASETS*N_VOLUMES
#define PD      8            // class period
#define INV_T   10.0f        // 1 / TEMP_FAC
#define KC      8192         // K-chunk per gram block (D_FEAT / 16)

typedef __attribute__((ext_vector_type(4)))  _Float16 v4h;
typedef __attribute__((ext_vector_type(8)))  _Float16 v8h;
typedef __attribute__((ext_vector_type(16))) _Float16 v16h;
typedef __attribute__((ext_vector_type(8)))  float    v8f;

#define LDS_AS __attribute__((address_space(3)))

// gfx1250 async global->LDS copy (16B per lane), tracked by ASYNCcnt.
__device__ __forceinline__ void async_ld_b128(void* lds_dst, const _Float16* gsrc) {
  unsigned off = (unsigned)(unsigned long long)(LDS_AS char*)lds_dst;
  asm volatile("global_load_async_to_lds_b128 %0, %1, off"
               :: "v"(off), "v"(gsrc) : "memory");
}
__device__ __forceinline__ void wait_async0() {
  asm volatile("s_wait_asynccnt 0x0" ::: "memory");
}

// ================= Kernel 1: per-row inverse norms =================
__global__ __launch_bounds__(256)
void rownorm_kernel(const float* __restrict__ X, float* __restrict__ inv_norm) {
  const int row = blockIdx.x;
  const float* xr = X + (size_t)row * D_FEAT;
  float ss = 0.f;
  for (int d = threadIdx.x * 4; d < D_FEAT; d += 256 * 4) {
    float4 v = *reinterpret_cast<const float4*>(xr + d);
    ss = fmaf(v.x, v.x, ss); ss = fmaf(v.y, v.y, ss);
    ss = fmaf(v.z, v.z, ss); ss = fmaf(v.w, v.w, ss);
  }
  __shared__ float red[256];
  red[threadIdx.x] = ss;
  __syncthreads();
  for (int s = 128; s > 0; s >>= 1) {
    if (threadIdx.x < s) red[threadIdx.x] += red[threadIdx.x + s];
    __syncthreads();
  }
  if (threadIdx.x == 0) {
    float n = fmaxf(sqrtf(red[0]), 1e-6f);   // torch eps clamp on the norm
    inv_norm[row] = 1.0f / n;
  }
}

// ========== Kernel 2: normalize rows and convert to f16 ==========
__global__ __launch_bounds__(256)
void normalize_f16_kernel(const float* __restrict__ X,
                          const float* __restrict__ inv_norm,
                          _Float16* __restrict__ Xh) {
  size_t idx = ((size_t)blockIdx.x * 256 + threadIdx.x) * 4;
  int row = (int)(idx >> 17);                // D_FEAT = 2^17
  float s = inv_norm[row];
  float4 v = *reinterpret_cast<const float4*>(X + idx);
  v4h h;
  h[0] = (_Float16)(v.x * s);
  h[1] = (_Float16)(v.y * s);
  h[2] = (_Float16)(v.z * s);
  h[3] = (_Float16)(v.w * s);
  *reinterpret_cast<v4h*>(Xh + idx) = h;
}

// ================= Kernel 3: zero the C accumulator =================
__global__ void zero_kernel(float* __restrict__ p, int n) {
  int i = blockIdx.x * blockDim.x + threadIdx.x;
  if (i < n) p[i] = 0.f;
}

// ========== Kernel 4: Gram matrix C += Xh * Xh^T via WMMA ==========
// Grid (6, 6, 16): 64x64 C tile per block, K split into 16 chunks of 8192.
// 8 waves / block; wave w -> row-tile (w&3), col-group (w>>2) of 32 cols.
// Double-buffered LDS fed by async global->LDS copies; K=64 per stage
// (4 independent wmma per workgroup barrier -> no D->C RAW hazard stalls).
__global__ __launch_bounds__(256)
void gram_wmma_kernel(const _Float16* __restrict__ Xh, float* __restrict__ C) {
  // 72-half row pitch (144 B): every b128 chunk stays 16B-aligned and the 16
  // fragment rows of a half-wave land on distinct bank groups (gcd(36,64)=4).
  __shared__ _Float16 As[2][64][72];
  __shared__ _Float16 Bs[2][64][72];

  const int i0    = blockIdx.x * 64;
  const int j0    = blockIdx.y * 64;
  const int kbase = blockIdx.z * KC;

  const int tid  = threadIdx.x;
  const int lrow = tid >> 2;             // 0..63 : LDS row this thread fills
  const int lseg = (tid & 3) * 8;        // f16 offset 0,8,16,24 (16B chunks)

  const _Float16* gA = Xh + (size_t)(i0 + lrow) * D_FEAT + kbase + lseg;
  const _Float16* gB = Xh + (size_t)(j0 + lrow) * D_FEAT + kbase + lseg;

  const int wave = tid >> 5;             // 0..7 (wave32)
  const int lane = tid & 31;
  const int half = lane >> 4;            // low/high half-wave
  const int l16  = lane & 15;
  const int r    = wave & 3;             // row tile   (16 rows)
  const int cg   = wave >> 2;            // col group  (32 cols)

  // independent accumulators per K sub-step: the 4 WMMAs in a stage have no
  // mutual dependence (avoids WMMA->WMMA RAW NOPs + acc register shuffles)
  v8f acc0[2] = {v8f{}, v8f{}};
  v8f acc1[2] = {v8f{}, v8f{}};

  // stage loader: 64 rows x 64 halfs per matrix, 4 async b128 per thread
  auto issue_stage = [&](int buf, int kk) {
    async_ld_b128(&As[buf][lrow][lseg],      gA + kk);
    async_ld_b128(&As[buf][lrow][lseg + 32], gA + kk + 32);
    async_ld_b128(&Bs[buf][lrow][lseg],      gB + kk);
    async_ld_b128(&Bs[buf][lrow][lseg + 32], gB + kk + 32);
  };

  const int NIT = KC / 64;               // 128 pipeline stages
  issue_stage(0, 0);

  for (int it = 0; it < NIT; ++it) {
    const int cur = it & 1;
    wait_async0();                       // this wave's copies into buf cur done
    __syncthreads();                     // all waves done: LDS cur valid,
                                         // buf cur^1 no longer being read
    if (it + 1 < NIT) issue_stage(cur ^ 1, (it + 1) * 64);

    const int h8  = half * 8;
    const int h16 = half * 16;
    const int arow  = r * 16 + l16;
    const int b0row = cg * 32 + l16;

#pragma unroll
    for (int ks = 0; ks < 2; ++ks) {     // two K=32 sub-steps
      const int ko = ks * 32;

      // A fragment (16x32 f16, ISA 7.12.2): low lanes K0-7|K16-23,
      // high lanes K8-15|K24-31, M = l16.
      v8h alo = *reinterpret_cast<const v8h*>(&As[cur][arow][ko + h8]);
      v8h ahi = *reinterpret_cast<const v8h*>(&As[cur][arow][ko + 16 + h8]);
      v16h a = __builtin_shufflevector(alo, ahi,
          0,1,2,3,4,5,6,7,8,9,10,11,12,13,14,15);

      // B fragment (32x16): lane = column N, low lanes K0-7|K16-23, high K8-15|K24-31.
      // For X*X^T a "column" of B is just row j of Xh -> same staging as A.
      v8h b0lo = *reinterpret_cast<const v8h*>(&Bs[cur][b0row][ko + h8]);
      v8h b0hi = *reinterpret_cast<const v8h*>(&Bs[cur][b0row][ko + 16 + h8]);
      v16h b0 = __builtin_shufflevector(b0lo, b0hi,
          0,1,2,3,4,5,6,7,8,9,10,11,12,13,14,15);
      v8h b1lo = *reinterpret_cast<const v8h*>(&Bs[cur][b0row + 16][ko + h8]);
      v8h b1hi = *reinterpret_cast<const v8h*>(&Bs[cur][b0row + 16][ko + 16 + h8]);
      v16h b1 = __builtin_shufflevector(b1lo, b1hi,
          0,1,2,3,4,5,6,7,8,9,10,11,12,13,14,15);

      acc0[ks] = __builtin_amdgcn_wmma_f32_16x16x32_f16(
          false, a, false, b0, (short)0, acc0[ks], false, false);
      acc1[ks] = __builtin_amdgcn_wmma_f32_16x16x32_f16(
          false, a, false, b1, (short)0, acc1[ks], false, false);
    }
  }

  // C/D layout: VGPR v holds M = v (lanes 0-15) / v+8 (lanes 16-31), N = l16.
  const int row0 = i0 + r * 16 + half * 8;
  const int col0 = j0 + cg * 32 + l16;
#pragma unroll
  for (int v = 0; v < 8; ++v) {
    atomicAdd(&C[(size_t)(row0 + v) * N_TOTAL + col0],
              acc0[0][v] + acc0[1][v]);
    atomicAdd(&C[(size_t)(row0 + v) * N_TOTAL + col0 + 16],
              acc1[0][v] + acc1[1][v]);
  }
}

// ========== Kernel 5: exp / class sums / pair loss (single block) ==========
__global__ __launch_bounds__(128)
void loss_kernel(const float* __restrict__ C, float* __restrict__ out) {
  __shared__ float rowsum_s[N_TOTAL];
  __shared__ float cls_s[N_TOTAL][PD];
  __shared__ float red[128];
  const int tid = threadIdx.x;

  // Phase 1: row sums + per-class (j mod 8) column sums of e = exp(10*sim).
  for (int rr = tid; rr < N_TOTAL; rr += 128) {
    const float* Crow = C + (size_t)rr * N_TOTAL;
    float rs = 0.f;
    float cs[PD];
#pragma unroll
    for (int c = 0; c < PD; ++c) cs[c] = 0.f;
    for (int j = 0; j < N_TOTAL; j += PD) {
#pragma unroll
      for (int c = 0; c < PD; ++c) {      // static c -> cs stays in VGPRs
        float e = __expf(INV_T * Crow[j + c]);
        rs += e;
        cs[c] += e;
      }
    }
    rowsum_s[rr] = rs;
#pragma unroll
    for (int c = 0; c < PD; ++c) cls_s[rr][c] = cs[c];
  }
  __syncthreads();

  // Phase 2: thread p handles pos_index p (0..127), 7 pair terms.
  const int p   = tid;
  const int rem = p & (PD - 1);
  const int i1 = p;
  const int i2 = BS + p;
  const int i3 = (p + PD) % N_TOTAL;
  const int i4 = (BS + p + PD) % N_TOTAL;
  const int i5 = 2 * BS + p;
  const int i6 = (2 * BS + p + PD) % N_TOTAL;

  auto pairLoss = [&](int a, int b) -> float {
    float en = __expf(INV_T * C[(size_t)a * N_TOTAL + b]);
    float da = rowsum_s[a] - cls_s[a][rem];   // excluded-class denominator
    float db = rowsum_s[b] - cls_s[b][rem];
    return -__logf(en / (en + da)) - __logf(en / (en + db));
  };

  float partial = pairLoss(i1, i2) + pairLoss(i3, i4) + pairLoss(i1, i3)
                + pairLoss(i2, i4) + pairLoss(i5, i6) + pairLoss(i2, i5)
                + pairLoss(i4, i6);

  red[tid] = partial;
  __syncthreads();
  for (int s = 64; s > 0; s >>= 1) {
    if (tid < s) red[tid] += red[tid + s];
    __syncthreads();
  }
  if (tid == 0) out[0] = red[0] / 768.0f;     // / (N_TRANSFORMS * 3 * BS)
}

// =========================== launcher ===========================
extern "C" void kernel_launch(void* const* d_in, const int* in_sizes, int n_in,
                              void* d_out, int out_size, void* d_ws, size_t ws_size,
                              hipStream_t stream) {
  (void)in_sizes; (void)n_in; (void)out_size; (void)ws_size;
  const float* X = (const float*)d_in[0];       // [384, 131072] fp32
  float* out = (float*)d_out;                   // scalar fp32

  // workspace layout: inv_norm | Xh (f16, 100.66 MB) | C (384x384 fp32)
  char* ws = (char*)d_ws;
  float*    inv_norm = (float*)ws;
  _Float16* Xh       = (_Float16*)(ws + 2048);
  float*    C        = (float*)(ws + 2048 + (size_t)N_TOTAL * D_FEAT * 2);

  rownorm_kernel<<<N_TOTAL, 256, 0, stream>>>(X, inv_norm);

  const size_t totalElems = (size_t)N_TOTAL * D_FEAT;      // 50,331,648
  const int cvtBlocks = (int)(totalElems / 4 / 256);       // 49152
  normalize_f16_kernel<<<cvtBlocks, 256, 0, stream>>>(X, inv_norm, Xh);

  const int cElems = N_TOTAL * N_TOTAL;
  zero_kernel<<<(cElems + 255) / 256, 256, 0, stream>>>(C, cElems);

  gram_wmma_kernel<<<dim3(6, 6, D_FEAT / KC), 256, 0, stream>>>(Xh, C);

  loss_kernel<<<1, 128, 0, stream>>>(C, out);
}
